// BiDirectionalSymplecticLayer_30717606101199
// MI455X (gfx1250) — matrix operand order repaired and checked by reading, not compiled, not run on hardware
//
#include <hip/hip_runtime.h>

#define AS1 __attribute__((address_space(1)))
#define AS3 __attribute__((address_space(3)))

typedef __attribute__((ext_vector_type(16))) _Float16 v16h;
typedef __attribute__((ext_vector_type(8)))  _Float16 v8h;
typedef __attribute__((ext_vector_type(8)))  float    v8f;
typedef __attribute__((ext_vector_type(4)))  float    v4f;

#define FDIM 128          // q / p feature dim
#define DDIM 256          // state dim = 2F
#define TSEQ 64
#define NT 16             // 256/16 output tiles
#define KT 8              // 256/32 k chunks
#define WAVES 4           // waves per block
#define NSTEPS 4          // leapfrog steps per direction
#define DTV 0.1f

#if __has_builtin(__builtin_amdgcn_tanhf)
#define TANHF(x) __builtin_amdgcn_tanhf(x)
#else
#define TANHF(x) tanhf(x)
#endif

// ---------------- weight prep: f32 -> f16, both orientations ----------------
__global__ void prep_weights(const float* __restrict__ W1g, const float* __restrict__ W2g,
                             _Float16* __restrict__ W1Tg, _Float16* __restrict__ W2Tg,
                             _Float16* __restrict__ W1Rg, _Float16* __restrict__ W2Rg) {
  const AS1 float* W1 = (const AS1 float*)W1g;
  const AS1 float* W2 = (const AS1 float*)W2g;
  AS1 _Float16* W1T = (AS1 _Float16*)W1Tg;
  AS1 _Float16* W2T = (AS1 _Float16*)W2Tg;
  AS1 _Float16* W1R = (AS1 _Float16*)W1Rg;
  AS1 _Float16* W2R = (AS1 _Float16*)W2Rg;
  int idx = blockIdx.x * blockDim.x + threadIdx.x;   // 0..65535
  int k = idx >> 8, n = idx & 255;
  float w1 = W1[idx], w2 = W2[idx];
  W1R[idx] = (_Float16)w1;            // row-major: B-operand for  du1 @ W1^T
  W2R[idx] = (_Float16)w2;            // row-major: B-operand for  g2  @ W2^T
  W1T[n * DDIM + k] = (_Float16)w1;   // transposed: B-operand for S  @ W1
  W2T[n * DDIM + k] = (_Float16)w2;   // transposed: B-operand for h1 @ W2
}

// ------------- fragment builders (CDNA5 16-bit WMMA VGPR layouts) -----------
// A (16x32): lane<16 holds K = k0+0..7 and k0+16..23 of row (lane&15); lane>=16: +8.
__device__ __forceinline__ v16h afrag_f32(const AS3 float* row, int k0) {
  v16h f;
#pragma unroll
  for (int j = 0; j < 8; ++j) {
    f[j]     = (_Float16)row[k0 + j];
    f[8 + j] = (_Float16)row[k0 + 16 + j];
  }
  return f;
}
__device__ __forceinline__ v16h afrag_f16(const AS3 _Float16* row, int k0) {
  v8h lo = *(const AS3 v8h*)(row + k0);
  v8h hi = *(const AS3 v8h*)(row + k0 + 16);
  return __builtin_shufflevector(lo, hi, 0,1,2,3,4,5,6,7,8,9,10,11,12,13,14,15);
}
// B (32x16): lane<16 holds K = kbase+0..7 and kbase+16..23 of column (lane&15); lane>=16: +8.
__device__ __forceinline__ v16h bfrag(const AS1 _Float16* wrow, int kbase) {
  v8h lo = *(const AS1 v8h*)(wrow + kbase);
  v8h hi = *(const AS1 v8h*)(wrow + kbase + 16);
  return __builtin_shufflevector(lo, hi, 0,1,2,3,4,5,6,7,8,9,10,11,12,13,14,15);
}

// 16-row x 16-col output tile: accumulate over K with double-buffered B fragments
__device__ __forceinline__ v8f mm_row(const v16h af[KT], const AS1 _Float16* wrow,
                                      int bkoff, v8f acc) {
  v16h bcur = bfrag(wrow, bkoff);
#pragma unroll
  for (int kt = 0; kt < KT; ++kt) {
    v16h bnext = (kt + 1 < KT) ? bfrag(wrow, (kt + 1) * 32 + bkoff) : bcur;
    acc = __builtin_amdgcn_wmma_f32_16x16x32_f16(false, af[kt], false, bcur,
                                                 (short)0, acc, false, false);
    bcur = bnext;
  }
  return acc;
}

// ------------------------- one gradient evaluation --------------------------
__device__ __forceinline__ void grad_eval(
    AS3 float* S, AS3 _Float16* H1, AS3 _Float16* G2,
    const AS1 _Float16* W1T, const AS1 _Float16* W2T,
    const AS1 _Float16* W1R, const AS1 _Float16* W2R,
    const AS1 float* b1, const AS1 float* b2, const AS1 float* wout,
    int lane, float pcoef, float qcoef, bool do_drift) {
  const int hi16  = (lane & 16);
  const int akoff = hi16 >> 1;   // 0 or 8  (A interleave)
  const int bkoff = hi16 >> 1;   // 0 or 8  (B interleave)
  const int moff  = hi16 >> 1;   // 0 or 8  (C row offset)
  const int ncol  = lane & 15;
  const int arow  = lane & 15;
  v16h af[KT];

  // ---- GEMM1: U1 = S @ W1 + b1 ; H1 = tanh(U1) ----
#pragma unroll
  for (int kt = 0; kt < KT; ++kt) af[kt] = afrag_f32(S + arow * DDIM, kt * 32 + akoff);
  for (int nt = 0; nt < NT; ++nt) {
    int n = nt * 16 + ncol;
    float bn = b1[n];
    v8f acc;
#pragma unroll
    for (int i = 0; i < 8; ++i) acc[i] = bn;
    acc = mm_row(af, W1T + (size_t)n * DDIM, bkoff, acc);
#pragma unroll
    for (int i = 0; i < 8; ++i) H1[(i + moff) * DDIM + n] = (_Float16)TANHF(acc[i]);
  }

  // ---- GEMM2: U2 = H1 @ W2 + b2 ; G2 = Wout * (1 - tanh(U2)^2) ----
#pragma unroll
  for (int kt = 0; kt < KT; ++kt) af[kt] = afrag_f16(H1 + arow * DDIM, kt * 32 + akoff);
  for (int nt = 0; nt < NT; ++nt) {
    int n = nt * 16 + ncol;
    float bn = b2[n];
    v8f acc;
#pragma unroll
    for (int i = 0; i < 8; ++i) acc[i] = bn;
    acc = mm_row(af, W2T + (size_t)n * DDIM, bkoff, acc);
    float wn = wout[n];
#pragma unroll
    for (int i = 0; i < 8; ++i) {
      float h2 = TANHF(acc[i]);
      G2[(i + moff) * DDIM + n] = (_Float16)(wn * (1.0f - h2 * h2));
    }
  }

  // ---- GEMM3: DH1 = G2 @ W2^T ; DU1 = DH1 * (1 - h1^2)  (overwrite G2) ----
#pragma unroll
  for (int kt = 0; kt < KT; ++kt) af[kt] = afrag_f16(G2 + arow * DDIM, kt * 32 + akoff);
  for (int nt = 0; nt < NT; ++nt) {
    int n = nt * 16 + ncol;
    v8f acc = {};
    acc = mm_row(af, W2R + (size_t)n * DDIM, bkoff, acc);
#pragma unroll
    for (int i = 0; i < 8; ++i) {
      int m = i + moff;
      float h1 = (float)H1[m * DDIM + n];
      G2[m * DDIM + n] = (_Float16)(acc[i] * (1.0f - h1 * h1));
    }
  }

  // ---- GEMM4: dS = DU1 @ W1^T ; fused leapfrog kicks into S ----
#pragma unroll
  for (int kt = 0; kt < KT; ++kt) af[kt] = afrag_f16(G2 + arow * DDIM, kt * 32 + akoff);
  for (int nt = 0; nt < NT; ++nt) {
    int n = nt * 16 + ncol;
    v8f acc = {};
    acc = mm_row(af, W1R + (size_t)n * DDIM, bkoff, acc);
#pragma unroll
    for (int i = 0; i < 8; ++i) {
      int m = i + moff;
      float g = acc[i];
      if (n < FDIM) {
        S[m * DDIM + FDIM + n] -= pcoef * g;       // p += pcoef * dp, dp = -g
      } else if (do_drift) {
        S[m * DDIM + (n - FDIM)] += qcoef * g;     // q += qcoef * dq, dq = g
      }
    }
  }
}

// ------------------------------ main kernel ---------------------------------
__global__ __launch_bounds__(WAVES * 32) void hnn_leapfrog(
    const float* __restrict__ xg,
    const float* __restrict__ b1g, const float* __restrict__ b2g,
    const float* __restrict__ woutg,
    const _Float16* __restrict__ W1Tg, const _Float16* __restrict__ W2Tg,
    const _Float16* __restrict__ W1Rg, const _Float16* __restrict__ W2Rg,
    float* __restrict__ outg) {
  __shared__ float    Ssh [WAVES * 16 * DDIM];   // 64 KB
  __shared__ _Float16 H1sh[WAVES * 16 * DDIM];   // 32 KB
  __shared__ _Float16 G2sh[WAVES * 16 * DDIM];   // 32 KB

  const AS1 float* x    = (const AS1 float*)xg;
  const AS1 float* b1   = (const AS1 float*)b1g;
  const AS1 float* b2   = (const AS1 float*)b2g;
  const AS1 float* wout = (const AS1 float*)woutg;
  const AS1 _Float16* W1T = (const AS1 _Float16*)W1Tg;
  const AS1 _Float16* W2T = (const AS1 _Float16*)W2Tg;
  const AS1 _Float16* W1R = (const AS1 _Float16*)W1Rg;
  const AS1 _Float16* W2R = (const AS1 _Float16*)W2Rg;
  AS1 float* out = (AS1 float*)outg;

  const int lane = threadIdx.x & 31;
  const int wave = threadIdx.x >> 5;
  AS3 float*    S  = (AS3 float*)   &Ssh [wave * 16 * DDIM];
  AS3 _Float16* H1 = (AS3 _Float16*)&H1sh[wave * 16 * DDIM];
  AS3 _Float16* G2 = (AS3 _Float16*)&G2sh[wave * 16 * DDIM];

  int gtile = blockIdx.x * WAVES + wave;        // 0..1023
  bool forward = (gtile < 512);
  int tile = forward ? gtile : gtile - 512;     // 0..511
  int row0 = tile * 16;
  float dt = forward ? DTV : -DTV;

  // init S = [q | p] from x[:,32,:] and x[:,31,:]
  for (int m = 0; m < 16; ++m) {
    const AS1 float* xq = x + ((size_t)(row0 + m) * TSEQ + 32) * FDIM;
    const AS1 float* xm = x + ((size_t)(row0 + m) * TSEQ + 31) * FDIM;
    v4f a = *(const AS1 v4f*)(xq + lane * 4);
    v4f b = *(const AS1 v4f*)(xm + lane * 4);
    v4f pv;
#pragma unroll
    for (int j = 0; j < 4; ++j) pv[j] = a[j] - b[j];
    *(AS3 v4f*)(S + m * DDIM + lane * 4)        = a;
    *(AS3 v4f*)(S + m * DDIM + FDIM + lane * 4) = pv;
  }

  for (int s = 0; s < NSTEPS; ++s) {
    grad_eval(S, H1, G2, W1T, W2T, W1R, W2R, b1, b2, wout, lane, 0.5f * dt, dt, true);
    grad_eval(S, H1, G2, W1T, W2T, W1R, W2R, b1, b2, wout, lane, 0.5f * dt, 0.0f, false);
  }

  // out layout per row: [q_b | p_b | q_mid | p_mid | q_f | p_f] (6 x 128)
  for (int m = 0; m < 16; ++m) {
    AS1 float* o = out + (size_t)(row0 + m) * 768;
    int c = lane * 4;
    v4f q = *(const AS3 v4f*)(S + m * DDIM + c);
    v4f p = *(const AS3 v4f*)(S + m * DDIM + FDIM + c);
    if (forward) {
      *(AS1 v4f*)(o + 512 + c) = q;
      *(AS1 v4f*)(o + 640 + c) = p;
      const AS1 float* xq = x + ((size_t)(row0 + m) * TSEQ + 32) * FDIM;
      const AS1 float* xm = x + ((size_t)(row0 + m) * TSEQ + 31) * FDIM;
      v4f a = *(const AS1 v4f*)(xq + c);
      v4f b = *(const AS1 v4f*)(xm + c);
      v4f pm;
#pragma unroll
      for (int j = 0; j < 4; ++j) pm[j] = a[j] - b[j];
      *(AS1 v4f*)(o + 256 + c) = a;
      *(AS1 v4f*)(o + 384 + c) = pm;
    } else {
      *(AS1 v4f*)(o + 0 + c)   = q;
      *(AS1 v4f*)(o + 128 + c) = p;
    }
  }
}

extern "C" void kernel_launch(void* const* d_in, const int* in_sizes, int n_in,
                              void* d_out, int out_size, void* d_ws, size_t ws_size,
                              hipStream_t stream) {
  const float* x    = (const float*)d_in[0];
  const float* W1   = (const float*)d_in[1];
  const float* b1   = (const float*)d_in[2];
  const float* W2   = (const float*)d_in[3];
  const float* b2   = (const float*)d_in[4];
  const float* Wout = (const float*)d_in[5];

  _Float16* W1T = (_Float16*)d_ws;
  _Float16* W2T = W1T + 256 * 256;
  _Float16* W1R = W2T + 256 * 256;
  _Float16* W2R = W1R + 256 * 256;

  prep_weights<<<256, 256, 0, stream>>>(W1, W2, W1T, W2T, W1R, W2R);

  // 1024 wave-tiles (512 fwd + 512 bwd), 4 waves / block, 32 KB LDS per wave
  hnn_leapfrog<<<256, WAVES * 32, 0, stream>>>(
      x, b1, b2, Wout, W1T, W2T, W1R, W2R, (float*)d_out);
}
